// SpaceTimeTransformer_30064771072827
// MI455X (gfx1250) — compile-verified, matrix operand order checked by reading, not run on hardware
//
#include <hip/hip_runtime.h>
#include <hip/hip_bf16.h>
#include <math.h>
#include <cstddef>

// ---------------------------------------------------------------------------
// SpaceTimeTransformer (TimeSformer divided space-time attention) forward.
// GEMMs run on the CDNA5 matrix pipe via v_wmma_f32_16x16x32_bf16 (bf16 in,
// f32 accumulate) with cooperative LDS staging of operand tiles.
// Attention (tiny per-head problems) stays on VALU.
// ---------------------------------------------------------------------------

typedef __attribute__((ext_vector_type(16))) __bf16 v16bf;
typedef __attribute__((ext_vector_type(8)))  float  v8f;

static constexpr int Bc   = 4;
static constexpr int Fc   = 8;
static constexpr int Hc   = 12;
static constexpr int Dc   = 768;
static constexpr int HIDc = 3072;
static constexpr int Nc   = 196;               // patches per frame (14x14)
static constexpr int Sc   = 1 + Fc * Nc;       // 1569 tokens
static constexpr int DH   = 64;                // head dim
static constexpr int ROWS = Bc * Sc;           // 6276 token rows

// GEMM macro-tile
static constexpr int BM = 64;
static constexpr int BN = 128;
static constexpr int BK = 32;

__device__ __forceinline__ float gelu_f(float v) {
  return 0.5f * v * (1.0f + erff(v * 0.70710678118654752f));
}

__device__ __forceinline__ v8f wmma_bf16(v16bf a, v16bf b, v8f c) {
  // D = A(16x32) * B(32x16) + C, f32 accumulate
  return __builtin_amdgcn_wmma_f32_16x16x32_bf16(false, a, false, b,
                                                 (short)0, c, false, false);
}

// Pack two f32 into one u32 of two bf16 (ISA operand granule).
__device__ __forceinline__ unsigned pack_bf16(float x, float y) {
  union { __bf16 h[2]; unsigned u; } cv;
  cv.h[0] = (__bf16)x;
  cv.h[1] = (__bf16)y;
  return cv.u;
}

// ---------------------------------------------------------------------------
// Generic GEMM:  C[M,N] = act( A[M,K] @ W[N,K]^T + bias[N] (+ res[M,N]) )
// Block = 256 threads (8 waves) computes a 64x128 C macro-tile:
//   wave grid 2(M) x 4(N); each wave a 32x32 tile = 2x2 WMMA 16x16 tiles.
// Per K-step of 32 the block cooperatively stages A(64x32) and B(128x32)
// into LDS as bf16, laid out in the ISA 7.12.2 WMMA operand element order:
//   uint index (row, half=lane>=16, ei2) holds K pair
//   kbase = (ei2<4 ? 2*ei2 : 2*ei2+8) + 8*half
// so each lane's v16bf fragment is 32 contiguous bytes in LDS.
// Requires N % 128 == 0, K % 32 == 0 (true for 768/2304/3072).
// ---------------------------------------------------------------------------
__global__ void __launch_bounds__(256)
gemm_bf16_kernel(const float* __restrict__ A, const float* __restrict__ W,
                 const float* __restrict__ bias, const float* __restrict__ res,
                 float* __restrict__ C, int M, int N, int K, int act) {
  __shared__ __align__(16) unsigned aLds[BM * 16];   // 4 KB
  __shared__ __align__(16) unsigned bLds[BN * 16];   // 8 KB

  const int tid  = threadIdx.x;
  const int lane = tid & 31;
  const int w    = tid >> 5;              // wave 0..7
  const int tilesN = N / BN;
  const int bm0 = (blockIdx.x / tilesN) * BM;
  const int bn0 = (blockIdx.x % tilesN) * BN;
  const int rl = lane & 15;               // row/col within 16-tile
  const int hh = lane >> 4;               // K-half selector
  const int wm = (w >> 2) * 32;           // wave tile origin in block
  const int wn = (w & 3) * 32;

  v8f acc[4];
  acc[0] = {}; acc[1] = {}; acc[2] = {}; acc[3] = {};

  for (int k0 = 0; k0 < K; k0 += BK) {
    // ---- stage A chunk: 64 rows x 32 K  -> 1024 uints, 4 per thread ----
#pragma unroll
    for (int i = 0; i < 4; ++i) {
      const int u    = tid + 256 * i;
      const int ei2  = u & 7;
      const int half = (u >> 3) & 1;
      const int m    = u >> 4;
      const int kb   = k0 + ((ei2 < 4) ? 2 * ei2 : 2 * ei2 + 8) + 8 * half;
      const int row  = min(bm0 + m, M - 1);          // clamp M tail
      const float* p = A + (size_t)row * K + kb;
      aLds[u] = pack_bf16(p[0], p[1]);
    }
    // ---- stage B chunk: 128 weight rows x 32 K -> 2048 uints, 8/thread ----
#pragma unroll
    for (int i = 0; i < 8; ++i) {
      const int u    = tid + 256 * i;
      const int ei2  = u & 7;
      const int half = (u >> 3) & 1;
      const int n    = u >> 4;
      const int kb   = k0 + ((ei2 < 4) ? 2 * ei2 : 2 * ei2 + 8) + 8 * half;
      const float* p = W + (size_t)(bn0 + n) * K + kb;
      bLds[u] = pack_bf16(p[0], p[1]);
    }
    __syncthreads();

    // ---- each wave reads its operand fragments from LDS (32B each) ----
    const v16bf a0 = *(const v16bf*)&aLds[((wm + rl)      * 2 + hh) * 8];
    const v16bf a1 = *(const v16bf*)&aLds[((wm + 16 + rl) * 2 + hh) * 8];
    const v16bf b0 = *(const v16bf*)&bLds[((wn + rl)      * 2 + hh) * 8];
    const v16bf b1 = *(const v16bf*)&bLds[((wn + 16 + rl) * 2 + hh) * 8];

    acc[0] = wmma_bf16(a0, b0, acc[0]);
    acc[1] = wmma_bf16(a0, b1, acc[1]);
    acc[2] = wmma_bf16(a1, b0, acc[2]);
    acc[3] = wmma_bf16(a1, b1, acc[3]);

    __syncthreads();   // protect LDS before next stage overwrites
  }

#pragma unroll
  for (int idx = 0; idx < 4; ++idx) {
    const int mt = idx >> 1, nt = idx & 1;
    const int ncol = bn0 + wn + nt * 16 + rl;
    const float bv = bias[ncol];
#pragma unroll
    for (int rr = 0; rr < 8; ++rr) {
      const int mrow = bm0 + wm + mt * 16 + rr + 8 * hh;  // ISA C/D f32 layout
      if (mrow < M) {
        float v = acc[idx][rr] + bv;
        if (res) v += res[(size_t)mrow * N + ncol];
        if (act) v = gelu_f(v);
        C[(size_t)mrow * N + ncol] = v;
      }
    }
  }
}

// ---------------------------------------------------------------------------
// im2col for patch embedding: A[rp, c] ; rp over B*F*196 patches, c over 768
// ---------------------------------------------------------------------------
__global__ void im2col_kernel(const float* __restrict__ x, float* __restrict__ A) {
  const int t = blockIdx.x * blockDim.x + threadIdx.x;
  const int total = Bc * Fc * Nc * Dc;
  if (t >= total) return;
  const int c  = t % Dc;
  const int rp = t / Dc;
  const int p  = rp % Nc;
  const int bf = rp / Nc;
  const int ph = p / 14, pw = p % 14;
  const int ch = c / 256, rem = c % 256;
  const int py = rem / 16, px = rem % 16;
  const size_t src = ((size_t)bf * 3 + ch) * 224 * 224 +
                     (size_t)(ph * 16 + py) * 224 + (pw * 16 + px);
  A[(size_t)rp * Dc + c] = x[src];
}

// ---------------------------------------------------------------------------
// Token assembly: xt[b,s,:] = patches + cls + positional + temporal embeds
// ---------------------------------------------------------------------------
__global__ void embed_kernel(const float* __restrict__ patch_out,
                             const float* __restrict__ cls_token,
                             const float* __restrict__ pos_embed,
                             const float* __restrict__ temporal_embed,
                             float* __restrict__ xt) {
  const int t = blockIdx.x * blockDim.x + threadIdx.x;
  const int total = Bc * Sc * Dc;
  if (t >= total) return;
  const int c = t % Dc;
  const int s = (t / Dc) % Sc;
  const int b = t / (Dc * Sc);
  float v;
  if (s == 0) {
    v = cls_token[c] + pos_embed[c];
  } else {
    const int j  = s - 1;
    const int fi = j / Nc, pi = j % Nc;
    v = patch_out[(((size_t)(b * Fc + fi)) * Nc + pi) * Dc + c] +
        pos_embed[(size_t)(1 + pi) * Dc + c] +
        temporal_embed[(size_t)fi * Dc + c];
  }
  xt[((size_t)b * Sc + s) * Dc + c] = v;
}

// ---------------------------------------------------------------------------
// LayerNorm: one wave32 per row of 768 (24 elems/lane, shuffle reductions)
// ---------------------------------------------------------------------------
__global__ void ln_kernel(const float* __restrict__ in, float* __restrict__ out,
                          const float* __restrict__ s, const float* __restrict__ b,
                          int rows, long ld_in, long ld_out) {
  const int wid  = (blockIdx.x * blockDim.x + threadIdx.x) >> 5;
  const int lane = threadIdx.x & 31;
  if (wid >= rows) return;
  const float* x = in + (size_t)wid * ld_in;
  float vals[24];
  float sum = 0.f;
#pragma unroll
  for (int i = 0; i < 24; ++i) { vals[i] = x[lane + 32 * i]; sum += vals[i]; }
#pragma unroll
  for (int off = 16; off; off >>= 1) sum += __shfl_xor(sum, off, 32);
  const float mean = sum * (1.0f / Dc);
  float vs = 0.f;
#pragma unroll
  for (int i = 0; i < 24; ++i) { float d = vals[i] - mean; vs += d * d; }
#pragma unroll
  for (int off = 16; off; off >>= 1) vs += __shfl_xor(vs, off, 32);
  const float rstd = rsqrtf(vs * (1.0f / Dc) + 1e-6f);
  float* o = out + (size_t)wid * ld_out;
#pragma unroll
  for (int i = 0; i < 24; ++i) {
    const int c = lane + 32 * i;
    o[c] = (vals[i] - mean) * rstd * s[c] + b[c];
  }
}

// qkv layout: qkv[b, s, col], col in [0,3D): q = col, k = D+col, v = 2D+col
// ---------------------------------------------------------------------------
// CLS attention: per (b,h), 1 query vs all S keys (online softmax).
// ---------------------------------------------------------------------------
__global__ void cls_attn_kernel(const float* __restrict__ qkv,
                                float* __restrict__ out) {
  const int t = blockIdx.x * blockDim.x + threadIdx.x;
  if (t >= Bc * Hc) return;
  const int b = t / Hc, h = t % Hc;
  const size_t row3 = 3 * Dc;
  const float* base = qkv + (size_t)b * Sc * row3;
  float q[DH];
  const float* qp = base + h * DH;
#pragma unroll
  for (int j = 0; j < DH; ++j) q[j] = qp[j] * 0.125f;
  float m = -1e30f, l = 0.f, acc[DH];
#pragma unroll
  for (int j = 0; j < DH; ++j) acc[j] = 0.f;
  for (int s = 0; s < Sc; ++s) {
    const float* kp = base + (size_t)s * row3 + Dc + h * DH;
    float dot = 0.f;
#pragma unroll
    for (int j = 0; j < DH; ++j) dot += q[j] * kp[j];
    const float mn = fmaxf(m, dot);
    const float sc = __expf(m - mn);
    const float p  = __expf(dot - mn);
    l = l * sc + p;
    const float* vp = base + (size_t)s * row3 + 2 * Dc + h * DH;
#pragma unroll
    for (int j = 0; j < DH; ++j) acc[j] = acc[j] * sc + p * vp[j];
    m = mn;
  }
  const float inv = 1.0f / l;
  float* op = out + (size_t)b * Sc * Dc + h * DH;   // token 0
#pragma unroll
  for (int j = 0; j < DH; ++j) op[j] = acc[j] * inv;
}

// ---------------------------------------------------------------------------
// Time attention: per (b,h,patch ni, frame fi) thread: keys = [cls] + same
// patch across 8 frames. 9-way softmax in registers.
// ---------------------------------------------------------------------------
__global__ void time_attn_kernel(const float* __restrict__ qkv,
                                 float* __restrict__ out) {
  int t = blockIdx.x * blockDim.x + threadIdx.x;
  if (t >= Bc * Hc * Nc * Fc) return;
  const int fi = t & 7;  t >>= 3;
  const int ni = t % Nc; t /= Nc;
  const int h  = t % Hc;
  const int b  = t / Hc;
  const size_t row3 = 3 * Dc;
  const float* base = qkv + (size_t)b * Sc * row3;
  const int sq = 1 + fi * Nc + ni;
  float q[DH];
  const float* qp = base + (size_t)sq * row3 + h * DH;
#pragma unroll
  for (int j = 0; j < DH; ++j) q[j] = qp[j] * 0.125f;

  float sc[Fc + 1];
#pragma unroll
  for (int kk = 0; kk <= Fc; ++kk) {
    const int sk = (kk == 0) ? 0 : (1 + (kk - 1) * Nc + ni);
    const float* kp = base + (size_t)sk * row3 + Dc + h * DH;
    float dot = 0.f;
#pragma unroll
    for (int j = 0; j < DH; ++j) dot += q[j] * kp[j];
    sc[kk] = dot;
  }
  float m = sc[0];
#pragma unroll
  for (int kk = 1; kk <= Fc; ++kk) m = fmaxf(m, sc[kk]);
  float l = 0.f;
#pragma unroll
  for (int kk = 0; kk <= Fc; ++kk) { sc[kk] = __expf(sc[kk] - m); l += sc[kk]; }
  const float inv = 1.0f / l;

  float acc[DH];
#pragma unroll
  for (int j = 0; j < DH; ++j) acc[j] = 0.f;
#pragma unroll
  for (int kk = 0; kk <= Fc; ++kk) {
    const int sk = (kk == 0) ? 0 : (1 + (kk - 1) * Nc + ni);
    const float* vp = base + (size_t)sk * row3 + 2 * Dc + h * DH;
    const float w = sc[kk];
#pragma unroll
    for (int j = 0; j < DH; ++j) acc[j] += w * vp[j];
  }
  float* op = out + ((size_t)b * Sc + sq) * Dc + h * DH;
#pragma unroll
  for (int j = 0; j < DH; ++j) op[j] = acc[j] * inv;
}

// ---------------------------------------------------------------------------
// Space attention: per (b,h,frame fi, query patch ni) thread: keys = [cls] +
// 196 patches of same frame. Online softmax (flash-style, per thread).
// ---------------------------------------------------------------------------
__global__ void space_attn_kernel(const float* __restrict__ qkv,
                                  float* __restrict__ out) {
  int t = blockIdx.x * blockDim.x + threadIdx.x;
  if (t >= Bc * Hc * Fc * Nc) return;
  const int ni = t % Nc; t /= Nc;
  const int fi = t & 7;  t >>= 3;
  const int h  = t % Hc;
  const int b  = t / Hc;
  const size_t row3 = 3 * Dc;
  const float* base = qkv + (size_t)b * Sc * row3;
  const int sq = 1 + fi * Nc + ni;
  float q[DH];
  const float* qp = base + (size_t)sq * row3 + h * DH;
#pragma unroll
  for (int j = 0; j < DH; ++j) q[j] = qp[j] * 0.125f;

  float m = -1e30f, l = 0.f, acc[DH];
#pragma unroll
  for (int j = 0; j < DH; ++j) acc[j] = 0.f;
  for (int kk = 0; kk <= Nc; ++kk) {
    const int sk = (kk == 0) ? 0 : (1 + fi * Nc + (kk - 1));
    const float* kp = base + (size_t)sk * row3 + Dc + h * DH;
    float dot = 0.f;
#pragma unroll
    for (int j = 0; j < DH; ++j) dot += q[j] * kp[j];
    const float mn = fmaxf(m, dot);
    const float scl = __expf(m - mn);
    const float p   = __expf(dot - mn);
    l = l * scl + p;
    const float* vp = base + (size_t)sk * row3 + 2 * Dc + h * DH;
#pragma unroll
    for (int j = 0; j < DH; ++j) acc[j] = acc[j] * scl + p * vp[j];
    m = mn;
  }
  const float inv = 1.0f / l;
  float* op = out + ((size_t)b * Sc + sq) * Dc + h * DH;
#pragma unroll
  for (int j = 0; j < DH; ++j) op[j] = acc[j] * inv;
}

// ---------------------------------------------------------------------------
// Host orchestration
// ---------------------------------------------------------------------------
static void gemm_launch(const float* A, const float* W, const float* bias,
                        const float* res, float* C, int M, int N, int K,
                        int act, hipStream_t stream) {
  const int tilesM = (M + BM - 1) / BM;
  const int tilesN = N / BN;
  gemm_bf16_kernel<<<tilesM * tilesN, 256, 0, stream>>>(A, W, bias, res, C,
                                                        M, N, K, act);
}

static void ln_launch(const float* in, float* out, const float* s,
                      const float* b, int rows, long ld_in, long ld_out,
                      hipStream_t stream) {
  const long threads = (long)rows * 32;
  const int blocks = (int)((threads + 255) / 256);
  ln_kernel<<<blocks, 256, 0, stream>>>(in, out, s, b, rows, ld_in, ld_out);
}

extern "C" void kernel_launch(void* const* d_in, const int* in_sizes, int n_in,
                              void* d_out, int out_size, void* d_ws, size_t ws_size,
                              hipStream_t stream) {
  (void)in_sizes; (void)n_in; (void)out_size; (void)ws_size;

  const float* x          = (const float*)d_in[0];
  const float* patch_w    = (const float*)d_in[1];
  const float* patch_b    = (const float*)d_in[2];
  const float* cls_token  = (const float*)d_in[3];
  const float* pos_embed  = (const float*)d_in[4];
  const float* temp_embed = (const float*)d_in[5];
  const float* norm1_s    = (const float*)d_in[6];
  const float* norm1_b    = (const float*)d_in[7];
  const float* norm2_s    = (const float*)d_in[8];
  const float* norm2_b    = (const float*)d_in[9];
  const float* norm3_s    = (const float*)d_in[10];
  const float* norm3_b    = (const float*)d_in[11];
  const float* attn_qkv_w = (const float*)d_in[12];
  const float* attn_qkv_b = (const float*)d_in[13];
  const float* attn_proj_w= (const float*)d_in[14];
  const float* attn_proj_b= (const float*)d_in[15];
  const float* time_qkv_w = (const float*)d_in[16];
  const float* time_qkv_b = (const float*)d_in[17];
  const float* time_proj_w= (const float*)d_in[18];
  const float* time_proj_b= (const float*)d_in[19];
  const float* fc1_w      = (const float*)d_in[20];
  const float* fc1_b      = (const float*)d_in[21];
  const float* fc2_w      = (const float*)d_in[22];
  const float* fc2_b      = (const float*)d_in[23];
  const float* norm_s     = (const float*)d_in[24];
  const float* norm_b     = (const float*)d_in[25];

  // Workspace layout (~174 MB):
  //   xt | y(ln) | attn_out | time_res | space_res | big (qkv OR mlp-hidden)
  char* ws = (char*)d_ws;
  const size_t SZ = (size_t)ROWS * Dc * sizeof(float);       // 19.28 MB
  float* xt   = (float*)(ws);
  float* y    = (float*)(ws + 1 * SZ);
  float* attn = (float*)(ws + 2 * SZ);
  float* res1 = (float*)(ws + 3 * SZ);
  float* res2 = (float*)(ws + 4 * SZ);
  float* big  = (float*)(ws + 5 * SZ);                       // 77.12 MB region

  // ---- patch embedding + token assembly ----
  {
    const int tot = Bc * Fc * Nc * Dc;
    im2col_kernel<<<(tot + 255) / 256, 256, 0, stream>>>(x, big);
    gemm_launch(big, patch_w, patch_b, nullptr, attn,
                Bc * Fc * Nc, Dc, Dc, 0, stream);
    const int tot2 = Bc * Sc * Dc;
    embed_kernel<<<(tot2 + 255) / 256, 256, 0, stream>>>(
        attn, cls_token, pos_embed, temp_embed, xt);
  }

  const int nAttnT = Bc * Hc * Nc * Fc;   // 75264 threads for both attn kernels

  for (int i = 0; i < 12; ++i) {
    // ---------------- time attention branch (norm3, time_qkv/proj) ---------
    ln_launch(xt, y, norm3_s + i * Dc, norm3_b + i * Dc, ROWS, Dc, Dc, stream);
    gemm_launch(y, time_qkv_w + (size_t)i * 3 * Dc * Dc,
                time_qkv_b + (size_t)i * 3 * Dc, nullptr, big,
                ROWS, 3 * Dc, Dc, 0, stream);
    cls_attn_kernel<<<1, 64, 0, stream>>>(big, attn);
    time_attn_kernel<<<(nAttnT + 255) / 256, 256, 0, stream>>>(big, attn);
    gemm_launch(attn, time_proj_w + (size_t)i * Dc * Dc,
                time_proj_b + (size_t)i * Dc, /*res=*/xt, res1,
                ROWS, Dc, Dc, 0, stream);

    // ---------------- space attention branch (norm1, attn_qkv/proj) --------
    ln_launch(res1, y, norm1_s + i * Dc, norm1_b + i * Dc, ROWS, Dc, Dc, stream);
    gemm_launch(y, attn_qkv_w + (size_t)i * 3 * Dc * Dc,
                attn_qkv_b + (size_t)i * 3 * Dc, nullptr, big,
                ROWS, 3 * Dc, Dc, 0, stream);
    cls_attn_kernel<<<1, 64, 0, stream>>>(big, attn);
    space_attn_kernel<<<(nAttnT + 255) / 256, 256, 0, stream>>>(big, attn);
    gemm_launch(attn, attn_proj_w + (size_t)i * Dc * Dc,
                attn_proj_b + (size_t)i * Dc, /*res=*/xt, res2,
                ROWS, Dc, Dc, 0, stream);

    // ---------------- MLP (norm2, fc1-gelu, fc2) ---------------------------
    ln_launch(res2, y, norm2_s + i * Dc, norm2_b + i * Dc, ROWS, Dc, Dc, stream);
    gemm_launch(y, fc1_w + (size_t)i * HIDc * Dc, fc1_b + (size_t)i * HIDc,
                nullptr, big, ROWS, HIDc, Dc, /*gelu=*/1, stream);
    gemm_launch(big, fc2_w + (size_t)i * Dc * HIDc, fc2_b + (size_t)i * Dc,
                /*res=*/res2, xt, ROWS, Dc, HIDc, 0, stream);
  }

  // ---- final LN on CLS token of each batch -> d_out [B, D] ----
  ln_launch(xt, (float*)d_out, norm_s, norm_b, Bc, (long)Sc * Dc, Dc, stream);
}
